// Model_39676907884336
// MI455X (gfx1250) — matrix operand order repaired and checked by reading, not compiled, not run on hardware
//
#include <hip/hip_runtime.h>

typedef __attribute__((ext_vector_type(16))) _Float16 v16h;
typedef __attribute__((ext_vector_type(8)))  _Float16 v8h;
typedef __attribute__((ext_vector_type(8)))  float    v8f;
typedef __attribute__((ext_vector_type(4)))  int      v4i;
typedef __attribute__((ext_vector_type(8)))  int      v8i;
typedef __attribute__((ext_vector_type(4)))  unsigned int v4u;

constexpr int BATCH = 16;
constexpr int SEQ   = 2048;
constexpr int DIM   = 128;
constexpr int WAVES = 8;     // waves per workgroup
constexpr int QT    = 16;    // query rows per wave
constexpr int KTILE = 32;    // keys per tile (= WMMA K for f16)

#define GAS __attribute__((address_space(1)))
#define LAS __attribute__((address_space(3)))

#if __has_builtin(__builtin_amdgcn_tensor_load_to_lds)
#define HAVE_TDM 1
#else
#define HAVE_TDM 0
#endif

#if __has_builtin(__builtin_amdgcn_global_load_async_to_lds_b128)
#define HAVE_ASYNC 1
#define ASYNC_CP16(dst, src) \
    __builtin_amdgcn_global_load_async_to_lds_b128((GAS v4i*)(src), (LAS v4i*)(dst), 0, 0)
#if __has_builtin(__builtin_amdgcn_s_wait_asynccnt)
#define WAIT_ASYNC(n) __builtin_amdgcn_s_wait_asynccnt(n)
#else
#define WAIT_ASYNC(n) asm volatile("s_wait_asynccnt %0" ::"i"(n) : "memory")
#endif
#else
#define HAVE_ASYNC 0
#define WAIT_ASYNC(n)
#endif

__device__ __forceinline__ unsigned lds_byte_offset(const void* p) {
    return (unsigned)(unsigned long long)(LAS const void*)p;
}

// ---------------- pre-pass: fp32 -> fp16 (and V transpose) ----------------
__global__ __launch_bounds__(256)
void cvt_qk_kernel(const float* __restrict__ Q, const float* __restrict__ K,
                   _Float16* __restrict__ Qh, _Float16* __restrict__ Kh)
{
    const size_t i = ((size_t)blockIdx.x * blockDim.x + threadIdx.x) * 8;
    float tq[8], tk[8];
    *(float4*)&tq[0] = ((const float4*)(Q + i))[0];
    *(float4*)&tq[4] = ((const float4*)(Q + i))[1];
    *(float4*)&tk[0] = ((const float4*)(K + i))[0];
    *(float4*)&tk[4] = ((const float4*)(K + i))[1];
    v8h oq, ok;
#pragma unroll
    for (int j = 0; j < 8; ++j) { oq[j] = (_Float16)tq[j]; ok[j] = (_Float16)tk[j]; }
    *(v8h*)(Qh + i) = oq;
    *(v8h*)(Kh + i) = ok;
}

__global__ __launch_bounds__(256)
void cvt_vt_kernel(const float* __restrict__ V, _Float16* __restrict__ Vt)
{
    // thread -> 8 consecutive s for fixed (b, d): Vt[b][d][s] = (f16) V[b][s][d]
    const size_t idx = (size_t)blockIdx.x * blockDim.x + threadIdx.x;
    const int spb = SEQ / 8;
    const int b   = idx / (DIM * spb);
    const int rem = idx % (DIM * spb);
    const int d   = rem / spb;
    const int s0  = (rem % spb) * 8;
    v8h o;
#pragma unroll
    for (int j = 0; j < 8; ++j)
        o[j] = (_Float16)V[(size_t)b * SEQ * DIM + (size_t)(s0 + j) * DIM + d];
    *(v8h*)&Vt[(size_t)b * DIM * SEQ + (size_t)d * SEQ + s0] = o;
}

// ---------------- main flash-attention kernel ----------------
__global__ __launch_bounds__(WAVES * 32)
void fa_wmma_kernel(const _Float16* __restrict__ Qh,
                    const _Float16* __restrict__ Kh,
                    const _Float16* __restrict__ Vt,
                    const float* __restrict__ Div,
                    float* __restrict__ Out)
{
    __shared__ _Float16 ldsK[2][KTILE * DIM];   // 2 x 8KB, [key][d]
    __shared__ _Float16 ldsV[2][DIM * KTILE];   // 2 x 8KB, [d][key]
    __shared__ _Float16 ldsP[WAVES][QT * KTILE];// 8KB, per-wave P bounce

    const int tid  = threadIdx.x;
    const int lane = tid & 31;
    const int wave = tid >> 5;
    const int g    = lane >> 4;
    const int n    = lane & 15;

    const int qchunks = SEQ / (WAVES * QT);
    const int b  = blockIdx.x / qchunks;
    const int q0 = (blockIdx.x % qchunks) * (WAVES * QT) + wave * QT;

    const _Float16* QhB = Qh + (size_t)b * SEQ * DIM;
    const _Float16* KhB = Kh + (size_t)b * SEQ * DIM;
    const _Float16* VtB = Vt + (size_t)b * DIM * SEQ;
    float*          Ob  = Out + (size_t)b * SEQ * DIM;

    // stage one K-tile + one Vt-tile into LDS
    auto stage_tile = [&](int kt, int buf) {
#if HAVE_TDM
        // Tensor Data Mover: one descriptor-driven DMA per matrix, issued by wave 0.
        if (wave == 0) {
            const v4i zero4 = {0, 0, 0, 0};
            const v8i zero8 = {0, 0, 0, 0, 0, 0, 0, 0};
            {   // K tile: 32 rows x 128 halves, row stride 128 (contiguous block)
                const unsigned long long ga =
                    (unsigned long long)(KhB + (size_t)kt * KTILE * DIM);
                v4u g0;
                g0.x = 1u;                                         // count=1 (valid D#)
                g0.y = lds_byte_offset(&ldsK[buf][0]);             // lds_addr (bytes)
                g0.z = (unsigned)ga;                               // global_addr[31:0]
                g0.w = ((unsigned)(ga >> 32) & 0x1FFFFFFu) | 0x80000000u; // [56:32] + type=2
                v8i g1 = { 0x10000,                 // data_size=1 (2 bytes)
                           (int)(128u << 16),       // tensor_dim0 = 128
                           (int)(2048u << 16),      // tensor_dim1 = 2048
                           (int)(128u << 16),       // tile_dim0 = 128
                           32,                      // tile_dim1 = 32
                           128,                     // tensor_dim0_stride = 128
                           0, 0 };
                __builtin_amdgcn_tensor_load_to_lds(g0, g1, zero4, zero4, zero8, 0);
            }
            {   // Vt tile: 128 rows x 32 halves, row stride SEQ
                const unsigned long long ga =
                    (unsigned long long)(VtB + (size_t)kt * KTILE);
                v4u g0;
                g0.x = 1u;
                g0.y = lds_byte_offset(&ldsV[buf][0]);
                g0.z = (unsigned)ga;
                g0.w = ((unsigned)(ga >> 32) & 0x1FFFFFFu) | 0x80000000u;
                v8i g1 = { 0x10000,                 // data_size=1 (2 bytes)
                           (int)(2048u << 16),      // tensor_dim0 = 2048
                           (int)(128u << 16),       // tensor_dim1 = 128
                           (int)(32u << 16),        // tile_dim0 = 32
                           128,                     // tile_dim1 = 128
                           2048,                    // tensor_dim0_stride = 2048
                           0, 0 };
                __builtin_amdgcn_tensor_load_to_lds(g0, g1, zero4, zero4, zero8, 0);
            }
        }
#else
        {   // K tile is a contiguous 32*128-half block; 16 halves per thread
            const _Float16* src = KhB + (size_t)kt * KTILE * DIM + tid * 16;
            _Float16*       dst = &ldsK[buf][tid * 16];
#if HAVE_ASYNC
            ASYNC_CP16(dst,     src);
            ASYNC_CP16(dst + 8, src + 8);
#else
            *(v8h*)dst       = *(const v8h*)src;
            *(v8h*)(dst + 8) = *(const v8h*)(src + 8);
#endif
        }
#pragma unroll
        for (int c = 0; c < 2; ++c) {   // Vt tile: 128 rows x 32 halves
            const int chunk = 2 * tid + c;     // 0..511
            const int row   = chunk >> 2;      // d row
            const int sub   = chunk & 3;       // 8-half sub-chunk
            const _Float16* src = VtB + (size_t)row * SEQ + (size_t)kt * KTILE + sub * 8;
            _Float16*       dst = &ldsV[buf][row * KTILE + sub * 8];
#if HAVE_ASYNC
            ASYNC_CP16(dst, src);
#else
            *(v8h*)dst = *(const v8h*)src;
#endif
        }
#endif
    };

    // ---- Q A-fragments (4 chunks of K=32), straight from f16 global ----
    v16h qf[4];
#pragma unroll
    for (int c = 0; c < 4; ++c) {
        v8h lo = *(const v8h*)&QhB[(q0 + n) * DIM + 32 * c + 16 * g];
        v8h hi = *(const v8h*)&QhB[(q0 + n) * DIM + 32 * c + 16 * g + 8];
        qf[c] = __builtin_shufflevector(lo, hi, 0,1,2,3,4,5,6,7,8,9,10,11,12,13,14,15);
    }

    v8f acc[8];
#pragma unroll
    for (int f = 0; f < 8; ++f) acc[f] = (v8f){};
    float mrow[8], lrow[8];
#pragma unroll
    for (int r = 0; r < 8; ++r) { mrow[r] = -__builtin_inff(); lrow[r] = 0.0f; }

    const int NT = SEQ / KTILE;   // 64
    stage_tile(0, 0);
    stage_tile(1, 1);

    for (int kt = 0; kt < NT; ++kt) {
        const int buf = kt & 1;
        const int kb  = kt * KTILE;

        // retire this tile's copies (tile kt+1 may stay in flight; none on last tile)
#if HAVE_TDM
        if (wave == 0) {
            if (kt + 1 < NT) __builtin_amdgcn_s_wait_tensorcnt(2);
            else             __builtin_amdgcn_s_wait_tensorcnt(0);
        }
#else
        if (kt + 1 < NT) { WAIT_ASYNC(4); } else { WAIT_ASYNC(0); }
#endif
        __syncthreads();     // all copies visible to all waves

        // ---- scores: two 16(q) x 16(key) fragments over D=128 ----
        v8f sc[2] = { (v8f){}, (v8f){} };
#pragma unroll
        for (int sub = 0; sub < 2; ++sub) {
#pragma unroll
            for (int c = 0; c < 4; ++c) {
                v16h bf = *(const v16h*)&ldsK[buf][(sub * 16 + n) * DIM + 32 * c + 16 * g];
                sc[sub] = __builtin_amdgcn_wmma_f32_16x16x32_f16(
                    false, qf[c], false, bf, (short)0, sc[sub], false, false);
            }
        }

        // ---- scale by 1/Div (v_rcp_f32), online softmax, stage P ----
#pragma unroll
        for (int r = 0; r < 8; ++r) {
            const int q = q0 + r + 8 * g;
            sc[0][r] *= __builtin_amdgcn_rcpf(Div[q * SEQ + kb + n]);
            sc[1][r] *= __builtin_amdgcn_rcpf(Div[q * SEQ + kb + 16 + n]);

            float rm = fmaxf(sc[0][r], sc[1][r]);
#pragma unroll
            for (int msk = 1; msk < 16; msk <<= 1)
                rm = fmaxf(rm, __shfl_xor(rm, msk, 32));

            const float mnew  = fmaxf(mrow[r], rm);
            const float scale = __expf(mrow[r] - mnew);
            const float p0 = __expf(sc[0][r] - mnew);
            const float p1 = __expf(sc[1][r] - mnew);

            const int row = r + 8 * g;
            ldsP[wave][row * KTILE + n]      = (_Float16)p0;
            ldsP[wave][row * KTILE + 16 + n] = (_Float16)p1;

            float rs = p0 + p1;
#pragma unroll
            for (int msk = 1; msk < 16; msk <<= 1)
                rs += __shfl_xor(rs, msk, 32);

            lrow[r] = lrow[r] * scale + rs;
            mrow[r] = mnew;
#pragma unroll
            for (int f = 0; f < 8; ++f) acc[f][r] *= scale;
        }

        // ---- reload P as A-fragment (DS in-order within wave) ----
        v8h plo = *(const v8h*)&ldsP[wave][n * KTILE + 16 * g];
        v8h phi = *(const v8h*)&ldsP[wave][n * KTILE + 16 * g + 8];
        v16h pf = __builtin_shufflevector(plo, phi, 0,1,2,3,4,5,6,7,8,9,10,11,12,13,14,15);

        // ---- O += P @ V ----
#pragma unroll
        for (int f = 0; f < 8; ++f) {
            v16h vf = *(const v16h*)&ldsV[buf][(16 * f + n) * KTILE + 16 * g];
            acc[f] = __builtin_amdgcn_wmma_f32_16x16x32_f16(
                false, pf, false, vf, (short)0, acc[f], false, false);
        }

        __syncthreads();     // everyone done with buf before overwrite
        if (kt + 2 < NT) stage_tile(kt + 2, buf);
    }

    // ---- normalize (one rcp per row) and store ----
    float linv[8];
#pragma unroll
    for (int r = 0; r < 8; ++r) linv[r] = __builtin_amdgcn_rcpf(lrow[r]);
#pragma unroll
    for (int f = 0; f < 8; ++f) {
#pragma unroll
        for (int r = 0; r < 8; ++r) {
            const int q = q0 + r + 8 * g;
            Ob[q * DIM + 16 * f + n] = acc[f][r] * linv[r];
        }
    }
}

extern "C" void kernel_launch(void* const* d_in, const int* in_sizes, int n_in,
                              void* d_out, int out_size, void* d_ws, size_t ws_size,
                              hipStream_t stream) {
    const float* Q  = (const float*)d_in[0];   // input1 [B,S,D]
    const float* K  = (const float*)d_in[1];   // input2 [B,S,D]
    const float* Dv = (const float*)d_in[2];   // input3 [S,S]
    const float* V  = (const float*)d_in[3];   // input4 [B,S,D]
    float* O = (float*)d_out;                  // [B,S,D]

    const size_t BSD = (size_t)BATCH * SEQ * DIM;
    _Float16* Qh = (_Float16*)d_ws;
    _Float16* Kh = Qh + BSD;
    _Float16* Vt = Kh + BSD;                   // [B, D, S] transposed

    const int cvt_blocks = (int)(BSD / 8 / 256);        // 2048
    hipLaunchKernelGGL(cvt_qk_kernel, dim3(cvt_blocks), dim3(256), 0, stream, Q, K, Qh, Kh);
    hipLaunchKernelGGL(cvt_vt_kernel, dim3(cvt_blocks), dim3(256), 0, stream, V, Vt);

    dim3 grid(BATCH * (SEQ / (WAVES * QT)));   // 256 workgroups
    dim3 block(WAVES * 32);                    // 8 wave32 waves
    hipLaunchKernelGGL(fa_wmma_kernel, grid, block, 0, stream, Qh, Kh, Vt, Dv, O);
}
